// StateGRNNEncoder_89215060672896
// MI455X (gfx1250) — matrix operand order repaired and checked by reading, not compiled, not run on hardware
//
#include <hip/hip_runtime.h>
#include <hip/hip_bf16.h>

// ---------------------------------------------------------------------------
// NRI encoder for MI455X (gfx1250, wave32, WMMA).
// All GEMMs run on v_wmma_f32_16x16x32_f16 (f16 in, f32 accumulate).
// Weights converted/prepacked to f16 once per call; activations kept f16;
// LSTM cell state f32. node2edge gathers are fused into the WMMA A-loader.
// ---------------------------------------------------------------------------

typedef __attribute__((ext_vector_type(16))) _Float16 v16h;
typedef __attribute__((ext_vector_type(8)))  float    v8f;
typedef __attribute__((ext_vector_type(4)))  float    f32x4;

#define Bc   8
#define Nc   16
#define Tc   64
#define Dc   8
#define Hc   256
#define Sc   4
#define Rc   240          // N*(N-1)
#define BTc  512          // B*T
#define BRc  1920         // B*R
#define BTNc 8192         // B*T*N
#define BTRc 122880       // B*T*R

// ---------------- WMMA fragment loaders (ISA 16-bit layouts) ----------------
// A (16x32, MxK, row-major source): lane m=lane&15, lg=lane>>4;
//   a[i] = A[m][ (i&7) + (i>>3)*16 + lg*8 ]  -> two contiguous 16B chunks.
__device__ __forceinline__ v16h ldA(const _Float16* __restrict__ At, int lda) {
  const int lane = threadIdx.x & 31;
  const int m = lane & 15, lg = lane >> 4;
  const _Float16* q = At + (size_t)m * lda + lg * 8;
  v16h a;
  *reinterpret_cast<f32x4*>(&a)       = *reinterpret_cast<const f32x4*>(q);
  *(reinterpret_cast<f32x4*>(&a) + 1) = *reinterpret_cast<const f32x4*>(q + 16);
  return a;
}
// B (32x16, KxN) where B[k][n] = W[n][k], W row-major (Nout,K):
//   lane n=lane&15, lg=lane>>4; b[i] = W[n][ (i&7) + (i>>3)*16 + lg*8 ] -> two 16B chunks.
__device__ __forceinline__ v16h ldB(const _Float16* __restrict__ Wt, int ldw) {
  const int lane = threadIdx.x & 31;
  const int n = lane & 15, lg = lane >> 4;
  const _Float16* q = Wt + (size_t)n * ldw + lg * 8;
  v16h b;
  *reinterpret_cast<f32x4*>(&b)       = *reinterpret_cast<const f32x4*>(q);
  *(reinterpret_cast<f32x4*>(&b) + 1) = *reinterpret_cast<const f32x4*>(q + 16);
  return b;
}

__device__ __forceinline__ float elu1(float x) {
  return x > 0.f ? x : (__expf(x) - 1.f);
}

// ---------------- generic GEMM: C_f16 = act(A_f16 @ W_f16^T + bias) ---------
__global__ __launch_bounds__(256) void k_gemm(
    const _Float16* __restrict__ A, int lda,
    const _Float16* __restrict__ W, int ldw,
    const float* __restrict__ bias,
    _Float16* __restrict__ C, int ldc,
    int M, int Nout, int K, int act)
{
  const int ntiles = Nout >> 4;
  const int wid = blockIdx.x * 8 + (threadIdx.x >> 5);
  const int tm = wid / ntiles, tn = wid % ntiles;
  if (tm * 16 >= M) return;                       // uniform per-wave
  const int lane = threadIdx.x & 31;
  const int n = lane & 15, mo = (lane >> 4) << 3;
  const float bv = bias[tn * 16 + n];
  v8f acc = {bv, bv, bv, bv, bv, bv, bv, bv};
  const _Float16* Ap = A + (size_t)tm * 16 * lda;
  const _Float16* Wp = W + (size_t)tn * 16 * ldw;
  for (int kt = 0; kt < K; kt += 32) {
    v16h a = ldA(Ap + kt, lda);
    v16h b = ldB(Wp + kt, ldw);
    acc = __builtin_amdgcn_wmma_f32_16x16x32_f16(false, a, false, b,
                                                 (short)0, acc, false, false);
  }
  _Float16* Cp = C + (size_t)(tm * 16 + mo) * ldc + tn * 16 + n;
#pragma unroll
  for (int v = 0; v < 8; ++v) {
    float x = acc[v];
    if (act) x = elu1(x);
    Cp[(size_t)v * ldc] = (_Float16)x;
  }
}

// --------- GEMM with node2edge gather fused into the A operand --------------
// XN: node features, rows (group*16 + node), 256 f16 cols (L2-resident).
// Logical A row m -> group g=m/240, edge r=m%240; K=512: k<256 -> sender row,
// k>=256 -> receiver row. Nout=256 fixed.
__global__ __launch_bounds__(256) void k_gemm_n2e(
    const _Float16* __restrict__ XN,
    const _Float16* __restrict__ W,      // (256,512) row-major f16
    const float* __restrict__ bias,
    _Float16* __restrict__ C, int ldc,
    int nGroups, int act)
{
  const int ntiles = 16;
  const int mtiles = nGroups * 15;       // 240/16 = 15
  const int wid = blockIdx.x * 8 + (threadIdx.x >> 5);
  const int tm = wid / ntiles, tn = wid % ntiles;
  if (tm >= mtiles) return;
  const int lane = threadIdx.x & 31;
  const int n = lane & 15, lg = lane >> 4, mo = lg << 3;
  const int ml = tm * 16 + n;            // A row handled by this lane
  const int g = ml / 240, r = ml % 240;
  const int snd = r / 15, j = r % 15;
  const int rcv = j + (j >= snd ? 1 : 0);
  const _Float16* rowS = XN + (size_t)(g * 16 + snd) * 256;
  const _Float16* rowR = XN + (size_t)(g * 16 + rcv) * 256;
  const float bv = bias[tn * 16 + n];
  v8f acc = {bv, bv, bv, bv, bv, bv, bv, bv};
  const _Float16* Wp = W + (size_t)tn * 16 * 512;
#pragma unroll
  for (int kt = 0; kt < 16; ++kt) {
    const _Float16* q = (kt < 8 ? rowS : rowR) + (kt & 7) * 32 + lg * 8;
    v16h a;
    *reinterpret_cast<f32x4*>(&a)       = *reinterpret_cast<const f32x4*>(q);
    *(reinterpret_cast<f32x4*>(&a) + 1) = *reinterpret_cast<const f32x4*>(q + 16);
    v16h b = ldB(Wp + kt * 32, 512);
    acc = __builtin_amdgcn_wmma_f32_16x16x32_f16(false, a, false, b,
                                                 (short)0, acc, false, false);
  }
  _Float16* Cp = C + (size_t)(tm * 16 + mo) * ldc + tn * 16 + n;
#pragma unroll
  for (int v = 0; v < 8; ++v) {
    float x = acc[v];
    if (act) x = elu1(x);
    Cp[(size_t)v * ldc] = (_Float16)x;
  }
}

// ---------------- BatchNorm (training-mode batch stats) ---------------------
__global__ __launch_bounds__(256) void k_bn_stats(
    const _Float16* __restrict__ X, int M, int C, float* __restrict__ stats)
{
  const int c = blockIdx.x, tid = threadIdx.x;
  float s = 0.f, s2 = 0.f;
  for (int m = tid; m < M; m += 256) {
    float v = (float)X[(size_t)m * C + c];
    s += v; s2 += v * v;
  }
  __shared__ float sh[512];
  sh[tid] = s; sh[tid + 256] = s2;
  __syncthreads();
  for (int o = 128; o > 0; o >>= 1) {
    if (tid < o) { sh[tid] += sh[tid + o]; sh[tid + 256] += sh[tid + 256 + o]; }
    __syncthreads();
  }
  if (tid == 0) {
    float mean = sh[0] / (float)M;
    float var  = sh[256] / (float)M - mean * mean;
    stats[c] = mean;
    stats[C + c] = rsqrtf(var + 1e-5f);
  }
}

__global__ void k_bn_apply(_Float16* X, const float* __restrict__ stats,
                           const float* __restrict__ g, const float* __restrict__ be,
                           int M, int C)
{
  long i = (long)blockIdx.x * blockDim.x + threadIdx.x;
  if (i >= (long)M * C) return;
  int c = (int)(i % C);
  float x = (float)X[i];
  X[i] = (_Float16)((x - stats[c]) * stats[C + c] * g[c] + be[c]);
}

// BN-apply for mlp2 output with (B,T,R,H)->(T,B,R,H) permute into XT.
__global__ void k_bn_apply_xt(const _Float16* __restrict__ X,
                              const float* __restrict__ stats,
                              const float* __restrict__ g, const float* __restrict__ be,
                              _Float16* __restrict__ XT)
{
  long i = (long)blockIdx.x * blockDim.x + threadIdx.x;
  if (i >= (long)BTRc * Hc) return;
  int c = (int)(i & 255);
  long m = i >> 8;
  int r  = (int)(m % Rc);
  int bt = (int)(m / Rc);
  int b  = bt >> 6, t = bt & 63;
  float x = (float)X[i];
  float y = (x - stats[c]) * stats[Hc + c] * g[c] + be[c];
  XT[(((size_t)(t * Bc + b) * Rc + r) << 8) + c] = (_Float16)y;
}

// ---------------- LSTM elementwise cell update ------------------------------
__global__ void k_lstm_ew(const _Float16* __restrict__ G, float* __restrict__ cst,
                          _Float16* __restrict__ H)
{
  long i = (long)blockIdx.x * blockDim.x + threadIdx.x;
  if (i >= (long)BRc * Hc) return;
  long m = i >> 8; int c = (int)(i & 255);
  const _Float16* g = G + m * 1024 + c;
  float ig = 1.f / (1.f + __expf(-(float)g[0]));
  float fg = 1.f / (1.f + __expf(-(float)g[256]));
  float gg = tanhf((float)g[512]);
  float og = 1.f / (1.f + __expf(-(float)g[768]));
  float cn = fg * cst[i] + ig * gg;
  cst[i] = cn;
  H[i] = (_Float16)(og * tanhf(cn));
}

// ---------------- edge2node (per-step): (B,R,H) -> (B*16,H), /16 ------------
__global__ void k_e2n(const _Float16* __restrict__ Hp, _Float16* __restrict__ EN)
{
  long i = (long)blockIdx.x * blockDim.x + threadIdx.x;
  if (i >= (long)Bc * 16 * 256) return;
  int c = (int)(i & 255), n = (int)((i >> 8) & 15), b = (int)(i >> 12);
  float s = 0.f;
#pragma unroll
  for (int snd = 0; snd < 16; ++snd) {
    if (snd == n) continue;
    int q = (n < snd) ? n : n - 1;                 // index of j=n in sender's list
    s += (float)Hp[((size_t)(b * Rc + snd * 15 + q) << 8) + c];
  }
  EN[i] = (_Float16)(s * 0.0625f);
}

// ---------------- final edge2node over concat(fwd,bwd) ----------------------
// ENF rows = (b*T+t)*16+n, 512 cols; OUTF/OUTB are (T,B,R,H) f16.
__global__ void k_e2n_final(const _Float16* __restrict__ OF,
                            const _Float16* __restrict__ OB,
                            _Float16* __restrict__ ENF)
{
  long i = (long)blockIdx.x * blockDim.x + threadIdx.x;
  if (i >= (long)BTNc * 512) return;
  int ch = (int)(i & 511);
  long row = i >> 9;
  int n  = (int)(row & 15);
  int bt = (int)(row >> 4);
  int b = bt >> 6, t = bt & 63;
  const _Float16* src = (ch < 256) ? OF : OB;
  int cc = ch & 255;
  size_t base = ((size_t)(t * Bc + b) * Rc) << 8;
  float s = 0.f;
#pragma unroll
  for (int snd = 0; snd < 16; ++snd) {
    if (snd == n) continue;
    int q = (n < snd) ? n : n - 1;
    s += (float)src[base + ((size_t)(snd * 15 + q) << 8) + cc];
  }
  ENF[i] = (_Float16)(s * 0.0625f);
}

// ---------------- small prep / util kernels ---------------------------------
__global__ void k_pad_mat(_Float16* dst, const float* __restrict__ src,
                          int dr, int dc, int sr, int sc)
{
  long i = (long)blockIdx.x * blockDim.x + threadIdx.x;
  if (i >= (long)dr * dc) return;
  int r = (int)(i / dc), c = (int)(i % dc);
  dst[i] = (r < sr && c < sc) ? (_Float16)src[(size_t)r * sc + c] : (_Float16)0.f;
}

__global__ void k_cat2(_Float16* dst, const float* __restrict__ s1, int k1,
                       const float* __restrict__ s2, int k2, int rows)
{
  const int K = k1 + k2;
  long i = (long)blockIdx.x * blockDim.x + threadIdx.x;
  if (i >= (long)rows * K) return;
  int r = (int)(i / K), c = (int)(i % K);
  dst[i] = (_Float16)(c < k1 ? s1[(size_t)r * k1 + c] : s2[(size_t)r * k2 + (c - k1)]);
}

__global__ void k_addvec(float* dst, const float* a, const float* b, int n)
{
  int i = blockIdx.x * blockDim.x + threadIdx.x;
  if (i < n) dst[i] = a[i] + b[i];
}

__global__ void k_padvec(float* dst, const float* src, int n, int sn)
{
  int i = blockIdx.x * blockDim.x + threadIdx.x;
  if (i < n) dst[i] = (i < sn) ? src[i] : 0.f;
}

// inputs (B,N,T,D) f32 -> XP (BT*16, 32) f16, K padded 8->32 with zeros.
__global__ void k_build_xp(const float* __restrict__ inp, _Float16* __restrict__ XP)
{
  long i = (long)blockIdx.x * blockDim.x + threadIdx.x;
  if (i >= (long)BTNc * 32) return;
  int c = (int)(i & 31);
  long row = i >> 5;
  int n  = (int)(row & 15);
  int bt = (int)(row >> 4);
  int b = bt >> 6, t = bt & 63;
  XP[i] = (c < Dc) ? (_Float16)inp[(((size_t)(b * Nc + n) * Tc) + t) * Dc + c]
                   : (_Float16)0.f;
}

__global__ void k_copy2d(_Float16* __restrict__ dst, int ldd,
                         const _Float16* __restrict__ src, int lds2,
                         int rows, int cols)
{
  long i = (long)blockIdx.x * blockDim.x + threadIdx.x;
  if (i >= (long)rows * cols) return;
  int r = (int)(i / cols), c = (int)(i % cols);
  dst[(size_t)r * ldd + c] = src[(size_t)r * lds2 + c];
}

// final BN + permute to output (B,N,T,S) f32.
__global__ void k_out(const _Float16* __restrict__ HO2, const float* __restrict__ stats,
                      const float* __restrict__ g, const float* __restrict__ be,
                      float* __restrict__ out)
{
  int i = blockIdx.x * blockDim.x + threadIdx.x;
  if (i >= Bc * Nc * Tc * Sc) return;
  int s = i & 3, t = (i >> 2) & 63, n = (i >> 8) & 15, b = i >> 12;
  int row = (b * Tc + t) * 16 + n;
  float v = (float)HO2[(size_t)row * 16 + s];
  out[i] = (v - stats[s]) * stats[16 + s] * g[s] + be[s];
}

// ---------------------------------------------------------------------------
extern "C" void kernel_launch(void* const* d_in, const int* in_sizes, int n_in,
                              void* d_out, int out_size, void* d_ws, size_t ws_size,
                              hipStream_t stream) {
  (void)in_sizes; (void)out_size;
  if (n_in < 53) return;

  const float* INP = (const float*)d_in[0];
  // mlp1
  const float *m1w1=(const float*)d_in[3], *m1b1=(const float*)d_in[4],
              *m1w2=(const float*)d_in[5], *m1b2=(const float*)d_in[6],
              *m1g =(const float*)d_in[7], *m1be=(const float*)d_in[8];
  // mlp2
  const float *m2w1=(const float*)d_in[9],  *m2b1=(const float*)d_in[10],
              *m2w2=(const float*)d_in[11], *m2b2=(const float*)d_in[12],
              *m2g =(const float*)d_in[13], *m2be=(const float*)d_in[14];
  // mlp3f / linearf / mlp3b / linearb
  const float *m3w1[2] = {(const float*)d_in[15], (const float*)d_in[23]};
  const float *m3b1[2] = {(const float*)d_in[16], (const float*)d_in[24]};
  const float *m3w2[2] = {(const float*)d_in[17], (const float*)d_in[25]};
  const float *m3b2[2] = {(const float*)d_in[18], (const float*)d_in[26]};
  const float *m3g [2] = {(const float*)d_in[19], (const float*)d_in[27]};
  const float *m3be[2] = {(const float*)d_in[20], (const float*)d_in[28]};
  const float *linw[2] = {(const float*)d_in[21], (const float*)d_in[29]};
  const float *linb[2] = {(const float*)d_in[22], (const float*)d_in[30]};
  // LSTMs: f0,f1,b0,b1  (wih, whh, bih, bhh)
  const float *wih[4], *whh[4], *bih[4], *bhh[4];
  for (int l = 0; l < 4; ++l) {
    wih[l] = (const float*)d_in[31 + 4 * l + 0];
    whh[l] = (const float*)d_in[31 + 4 * l + 1];
    bih[l] = (const float*)d_in[31 + 4 * l + 2];
    bhh[l] = (const float*)d_in[31 + 4 * l + 3];
  }
  // mlp_out
  const float *mow1=(const float*)d_in[47], *mob1=(const float*)d_in[48],
              *mow2=(const float*)d_in[49], *mob2=(const float*)d_in[50],
              *mog =(const float*)d_in[51], *mobe=(const float*)d_in[52];

  // ---------------- workspace layout ----------------
  size_t off = 0;
  auto ah = [&](size_t n) -> _Float16* {
    _Float16* r = (_Float16*)((char*)d_ws + off);
    off = (off + n * sizeof(_Float16) + 255) & ~(size_t)255; return r;
  };
  auto af = [&](size_t n) -> float* {
    float* r = (float*)((char*)d_ws + off);
    off = (off + n * sizeof(float) + 255) & ~(size_t)255; return r;
  };

  _Float16* W1P1 = ah(256 * 32);          // mlp1 fc1, K padded to 32
  _Float16* W21  = ah(256 * 256);
  _Float16* W12  = ah(256 * 512);
  _Float16* W22  = ah(256 * 256);
  _Float16* W13[2] = { ah(256 * 256), ah(256 * 256) };
  _Float16* W23[2] = { ah(256 * 256), ah(256 * 256) };
  _Float16* WLN[2] = { ah(256 * 512), ah(256 * 512) };
  _Float16* WC[4];                        // [wih|whh] cat: f0,f1,b0,b1
  WC[0] = ah(1024 * 512); WC[1] = ah(1024 * 768);
  WC[2] = ah(1024 * 512); WC[3] = ah(1024 * 768);
  _Float16* WO1 = ah(256 * 512);
  _Float16* WO2 = ah(16 * 256);           // mlp_out fc2, Nout padded 4->16
  float* BS[4]; for (int l = 0; l < 4; ++l) BS[l] = af(1024);
  float* BO2   = af(16);
  float* STATS = af(512);
  float* C0 = af((size_t)BRc * Hc);
  float* C1 = af((size_t)BRc * Hc);

  _Float16* XP   = ah((size_t)BTNc * 32);
  _Float16* HA   = ah((size_t)BTNc * Hc);
  _Float16* HB   = ah((size_t)BTNc * Hc);       // becomes X1 after in-place BN
  _Float16* BIG1 = ah((size_t)BTRc * Hc);       // mlp2 fc1 out; reused as OUTF
  _Float16* BIG2 = ah((size_t)BTRc * Hc);       // mlp2 fc2 out; reused as OUTB
  _Float16* XT   = ah((size_t)Tc * BRc * Hc);   // (T,B,R,H) f16
  _Float16* Z    = ah((size_t)BRc * 768);       // [x | h0 | h1]
  _Float16* GT   = ah((size_t)BRc * 1024);      // gates
  _Float16* HPRE = ah((size_t)BRc * Hc);
  _Float16* EN   = ah(128 * 256);
  _Float16* T1   = ah(128 * 256);
  _Float16* T2   = ah(128 * 256);
  _Float16* ENF  = ah((size_t)BTNc * 512);
  _Float16* HO1  = ah((size_t)BTNc * Hc);
  _Float16* HO2  = ah((size_t)BTNc * 16);
  if (off > ws_size) return;                    // workspace too small: bail

  _Float16* OUTF = BIG1;                        // (T,B,R,H), reuse after mlp2
  _Float16* OUTB = BIG2;

  auto eg = [](long n) { return (unsigned)((n + 255) / 256); };
  auto gemm = [&](const _Float16* A, int lda, const _Float16* W, int ldw,
                  const float* bias, _Float16* C, int ldc,
                  int M, int Nout, int K, int act) {
    long waves = (long)(M / 16) * (Nout / 16);
    unsigned blocks = (unsigned)((waves + 7) / 8);
    k_gemm<<<blocks, 256, 0, stream>>>(A, lda, W, ldw, bias, C, ldc, M, Nout, K, act);
  };
  auto gemm_n2e = [&](const _Float16* XN, const _Float16* W, const float* bias,
                      _Float16* C, int ldc, int nGroups, int act) {
    long waves = (long)nGroups * 15 * 16;
    unsigned blocks = (unsigned)((waves + 7) / 8);
    k_gemm_n2e<<<blocks, 256, 0, stream>>>(XN, W, bias, C, ldc, nGroups, act);
  };

  // ---------------- weight prep (f32 -> f16, pad/concat) ----------------
  k_pad_mat<<<eg(256 * 32), 256, 0, stream>>>(W1P1, m1w1, 256, 32, 256, 8);
  k_pad_mat<<<eg(256 * 256), 256, 0, stream>>>(W21, m1w2, 256, 256, 256, 256);
  k_pad_mat<<<eg(256 * 512), 256, 0, stream>>>(W12, m2w1, 256, 512, 256, 512);
  k_pad_mat<<<eg(256 * 256), 256, 0, stream>>>(W22, m2w2, 256, 256, 256, 256);
  for (int d = 0; d < 2; ++d) {
    k_pad_mat<<<eg(256 * 256), 256, 0, stream>>>(W13[d], m3w1[d], 256, 256, 256, 256);
    k_pad_mat<<<eg(256 * 256), 256, 0, stream>>>(W23[d], m3w2[d], 256, 256, 256, 256);
    k_pad_mat<<<eg(256 * 512), 256, 0, stream>>>(WLN[d], linw[d], 256, 512, 256, 512);
  }
  k_cat2<<<eg(1024 * 512), 256, 0, stream>>>(WC[0], wih[0], 256, whh[0], 256, 1024);
  k_cat2<<<eg(1024 * 768), 256, 0, stream>>>(WC[1], wih[1], 512, whh[1], 256, 1024);
  k_cat2<<<eg(1024 * 512), 256, 0, stream>>>(WC[2], wih[2], 256, whh[2], 256, 1024);
  k_cat2<<<eg(1024 * 768), 256, 0, stream>>>(WC[3], wih[3], 512, whh[3], 256, 1024);
  for (int l = 0; l < 4; ++l)
    k_addvec<<<eg(1024), 256, 0, stream>>>(BS[l], bih[l], bhh[l], 1024);
  k_pad_mat<<<eg(16 * 256), 256, 0, stream>>>(WO2, mow2, 16, 256, 4, 256);
  k_padvec<<<1, 256, 0, stream>>>(BO2, mob2, 16, 4);
  k_build_xp<<<eg((long)BTNc * 32), 256, 0, stream>>>(INP, XP);

  // ---------------- mlp1 (M=8192) ----------------
  gemm(XP, 32, W1P1, 32, m1b1, HA, 256, BTNc, 256, 32, 1);
  gemm(HA, 256, W21, 256, m1b2, HB, 256, BTNc, 256, 256, 1);
  k_bn_stats<<<256, 256, 0, stream>>>(HB, BTNc, 256, STATS);
  k_bn_apply<<<eg((long)BTNc * 256), 256, 0, stream>>>(HB, STATS, m1g, m1be, BTNc, 256);

  // ---------------- mlp2 over node2edge (M=122880) ----------------
  gemm_n2e(HB, W12, m2b1, BIG1, 256, BTc, 1);
  gemm(BIG1, 256, W22, 256, m2b2, BIG2, 256, BTRc, 256, 256, 1);
  k_bn_stats<<<256, 256, 0, stream>>>(BIG2, BTRc, 256, STATS);
  k_bn_apply_xt<<<eg((long)BTRc * 256), 256, 0, stream>>>(BIG2, STATS, m2g, m2be, XT);

  // ---------------- bidirectional graph-LSTM ----------------
  for (int dir = 0; dir < 2; ++dir) {
    const _Float16* W0 = WC[dir == 0 ? 0 : 2];
    const _Float16* W1 = WC[dir == 0 ? 1 : 3];
    const float* bs0 = BS[dir == 0 ? 0 : 2];
    const float* bs1 = BS[dir == 0 ? 1 : 3];
    _Float16* OUT = (dir == 0) ? OUTF : OUTB;

    hipMemsetAsync(Z,  0, (size_t)BRc * 768 * sizeof(_Float16), stream);
    hipMemsetAsync(C0, 0, (size_t)BRc * Hc * sizeof(float), stream);
    hipMemsetAsync(C1, 0, (size_t)BRc * Hc * sizeof(float), stream);

    auto graph_mix = [&](_Float16* dstSlot) {
      k_e2n<<<eg((long)Bc * 16 * 256), 256, 0, stream>>>(HPRE, EN);
      gemm(EN, 256, W13[dir], 256, m3b1[dir], T1, 256, 128, 256, 256, 1);
      gemm(T1, 256, W23[dir], 256, m3b2[dir], T2, 256, 128, 256, 256, 1);
      k_bn_stats<<<256, 256, 0, stream>>>(T2, 128, 256, STATS);
      k_bn_apply<<<eg(128 * 256), 256, 0, stream>>>(T2, STATS, m3g[dir], m3be[dir], 128, 256);
      gemm_n2e(T2, WLN[dir], linb[dir], dstSlot, 768, Bc, 0);
    };

    for (int ti = 0; ti < Tc; ++ti) {
      int t = dir ? (Tc - 1 - ti) : ti;
      // x_t into Z[:,0:256)
      k_copy2d<<<eg((long)BRc * 256), 256, 0, stream>>>(
          Z, 768, XT + (size_t)t * BRc * Hc, 256, BRc, 256);
      // cell0: gates over [x | h0_prev]
      gemm(Z, 768, W0, 512, bs0, GT, 1024, BRc, 1024, 512, 0);
      k_lstm_ew<<<eg((long)BRc * 256), 256, 0, stream>>>(GT, C0, HPRE);
      graph_mix(Z + 256);                           // h0_mixed -> Z[:,256:512)
      // cell1: gates over [x | h0_mixed | h1_prev]
      gemm(Z, 768, W1, 768, bs1, GT, 1024, BRc, 1024, 768, 0);
      k_lstm_ew<<<eg((long)BRc * 256), 256, 0, stream>>>(GT, C1, HPRE);
      graph_mix(Z + 512);                           // h1_mixed -> Z[:,512:768)
      // emit h1_mixed at time t
      k_copy2d<<<eg((long)BRc * 256), 256, 0, stream>>>(
          OUT + (size_t)t * BRc * Hc, 256, Z + 512, 768, BRc, 256);
    }
  }

  // ---------------- final: edge2node over concat + mlp_out ----------------
  k_e2n_final<<<eg((long)BTNc * 512), 256, 0, stream>>>(OUTF, OUTB, ENF);
  gemm(ENF, 512, WO1, 512, mob1, HO1, 256, BTNc, 256, 512, 1);
  gemm(HO1, 256, WO2, 256, BO2, HO2, 16, BTNc, 16, 256, 1);
  k_bn_stats<<<16, 256, 0, stream>>>(HO2, BTNc, 16, STATS);
  k_out<<<eg(Bc * Nc * Tc * Sc), 256, 0, stream>>>(HO2, STATS, mog, mobe, (float*)d_out);
}
